// MMSelfAttn_20057497272707
// MI455X (gfx1250) — matrix operand order repaired and checked by reading, not compiled, not run on hardware
//
#include <hip/hip_runtime.h>

// ---------------------------------------------------------------------------
// Flash-attention style multimodal self-attention for MI455X (gfx1250).
//   - bf16 WMMA (v_wmma_f32_16x16x32_bf16), fp32 softmax/accumulation
//   - prepass converts K/V to bf16 (V transposed) in d_ws; hot loop stages
//     tiles with double-buffered global_load_async_to_lds_b128 + asynccnt
//   - row-sum via P x ones WMMA; row-max via v_permlane16_b32 butterflies
// Fixed problem shape from setup_inputs(): B=2, S=4096, N=16, H=64, GS=128.
// ---------------------------------------------------------------------------

typedef __attribute__((ext_vector_type(16))) __bf16 bf16x16;
typedef __attribute__((ext_vector_type(8)))  __bf16 bf16x8;
typedef __attribute__((ext_vector_type(2)))  __bf16 bf16x2;
typedef __attribute__((ext_vector_type(8)))  float  v8f;
typedef __attribute__((ext_vector_type(4)))  unsigned int u32x4;

#define KPAD 72   // shorts per K-tile row   (144 B: 16B-aligned, bank stride 36)
#define VPAD 56   // shorts per Vt-tile row  (112 B: 16B-aligned, bank stride 28)
#define PPAD 56   // shorts per P-tile row

static __device__ __forceinline__ unsigned short bfbits(float f) {
  return __builtin_bit_cast(unsigned short, (__bf16)f);
}
static __device__ __forceinline__ unsigned pk2(float a, float b) {
  bf16x2 p; p[0] = (__bf16)a; p[1] = (__bf16)b;
  return __builtin_bit_cast(unsigned, p);
}
static __device__ __forceinline__ u32x4 pk8(const float* f) {
  u32x4 r;
  r[0] = pk2(f[0], f[1]); r[1] = pk2(f[2], f[3]);
  r[2] = pk2(f[4], f[5]); r[3] = pk2(f[6], f[7]);
  return r;
}

// Build a 16-element bf16 fragment from two 16-byte LDS groups.
static __device__ __forceinline__ bf16x16 frag2(const unsigned short* p, int lo, int hi) {
  bf16x8 a = *(const bf16x8*)(p + lo);
  bf16x8 b = *(const bf16x8*)(p + hi);
  bf16x16 r;
#pragma unroll
  for (int i = 0; i < 8; ++i) { r[i] = a[i]; r[i + 8] = b[i]; }
  return r;
}

static __device__ __forceinline__ v8f wmma_bf16(bf16x16 a, bf16x16 b, v8f c) {
  return __builtin_amdgcn_wmma_f32_16x16x32_bf16(false, a, false, b, (short)0, c,
                                                 false, false);
}

// xor-butterfly max across each 16-lane group via v_permlane16_b32 (VALU, no DS).
static __device__ __forceinline__ float pl16(float v, unsigned s0, unsigned s1) {
  unsigned x = __builtin_bit_cast(unsigned, v);
  x = __builtin_amdgcn_permlane16(x, x, s0, s1, false, false);
  return __builtin_bit_cast(float, x);
}
static __device__ __forceinline__ float rowmax16(float v) {
  v = fmaxf(v, pl16(v, 0x67452301u, 0xefcdab89u));  // xor 1
  v = fmaxf(v, pl16(v, 0x54761032u, 0xdcfe98bau));  // xor 2
  v = fmaxf(v, pl16(v, 0x32107654u, 0xba98fedcu));  // xor 4
  v = fmaxf(v, pl16(v, 0xfedcba98u, 0x76543210u));  // xor 8
  return v;
}

// CDNA5 async global->LDS copy (16 B per lane) + waits (CDNA5 ISA).
static __device__ __forceinline__ unsigned lds_off32(const void* p) {
  return (unsigned)(size_t)p;  // low 32 bits of LDS-aperture address = LDS offset
}
static __device__ __forceinline__ void async_cp16(unsigned lds, const void* g) {
  asm volatile("global_load_async_to_lds_b128 %0, %1, off"
               :: "v"(lds), "v"(g) : "memory");
}
static __device__ __forceinline__ void async_wait0() {
  asm volatile("s_wait_asynccnt 0x0" ::: "memory");
}
static __device__ __forceinline__ void async_wait2() {
  asm volatile("s_wait_asynccnt 0x2" ::: "memory");
}

// ---------------------------------------------------------------------------
// Prepass 1: convert f32 rows of H=64 -> bf16 rows, output (B,N,rows,H) packed.
// ---------------------------------------------------------------------------
__global__ __launch_bounds__(256) void cvt_rows_kernel(
    const float* __restrict__ in, unsigned short* __restrict__ out,
    int rows, size_t b_str, size_t n_str, size_t row_str, int N) {
  const int rbp = rows / 32;
  const int bid = blockIdx.x;
  const int b = bid / (N * rbp);
  const int rem = bid % (N * rbp);
  const int n = rem / rbp;
  const int row = (rem % rbp) * 32 + (threadIdx.x >> 3);
  const int g = threadIdx.x & 7;
  const float* src = in + b * b_str + n * n_str + (size_t)row * row_str + g * 8;
  float4 f0 = *(const float4*)src;
  float4 f1 = *(const float4*)(src + 4);
  float tf[8] = {f0.x, f0.y, f0.z, f0.w, f1.x, f1.y, f1.z, f1.w};
  unsigned short* dst = out + (((size_t)b * N + n) * rows + row) * 64 + g * 8;
  *(u32x4*)dst = pk8(tf);
}

// ---------------------------------------------------------------------------
// Prepass 2: f32 (rows,64) -> bf16 transposed (64,rows), per (b,n), 64-row tiles.
// ---------------------------------------------------------------------------
__global__ __launch_bounds__(256) void tr64_kernel(
    const float* __restrict__ in, unsigned short* __restrict__ out,
    int rows, size_t b_str, size_t n_str, size_t row_str, int N) {
  __shared__ __align__(16) unsigned short T[64 * 72];
  const int tbp = rows / 64;
  const int bid = blockIdx.x;
  const int b = bid / (N * tbp);
  const int rem = bid % (N * tbp);
  const int n = rem / tbp;
  const int s0 = (rem % tbp) * 64;
  const int t = threadIdx.x;
  {
    const int sr = t >> 2, hg = t & 3;
    const float* src = in + b * b_str + n * n_str + (size_t)(s0 + sr) * row_str + hg * 16;
    float tf[16];
#pragma unroll
    for (int i = 0; i < 16; i += 4) {
      float4 f = *(const float4*)(src + i);
      tf[i] = f.x; tf[i + 1] = f.y; tf[i + 2] = f.z; tf[i + 3] = f.w;
    }
    *(u32x4*)&T[sr * 72 + hg * 16]     = pk8(tf);
    *(u32x4*)&T[sr * 72 + hg * 16 + 8] = pk8(tf + 8);
  }
  __syncthreads();
  {
    const int h = t >> 2, sg = t & 3;
    __align__(16) unsigned short o[16];
#pragma unroll
    for (int i = 0; i < 16; ++i) o[i] = T[(sg * 16 + i) * 72 + h];
    unsigned short* dst = out + (((size_t)b * N + n) * 64 + h) * (size_t)rows + s0 + sg * 16;
    *(u32x4*)&dst[0] = *(u32x4*)&o[0];
    *(u32x4*)&dst[8] = *(u32x4*)&o[8];
  }
}

// ---------------------------------------------------------------------------
// Main attention kernel. PRE=1: double-buffered async staging from d_ws.
// PRE=0: fallback, stage+convert f32 sources through VGPRs (no d_ws needed).
// ---------------------------------------------------------------------------
template <bool PRE>
__global__ __launch_bounds__(256) void mmattn_fa_kernel(
    const long long* __restrict__ modal_index, int nseg,
    const float* __restrict__ Q, const float* __restrict__ K,
    const float* __restrict__ V, const float* __restrict__ lmask,
    const int* __restrict__ gflag, const float* __restrict__ GK,
    const float* __restrict__ GV, const float* __restrict__ gmask,
    const unsigned short* __restrict__ Kp, const unsigned short* __restrict__ Vp,
    const unsigned short* __restrict__ GKp, const unsigned short* __restrict__ GVp,
    float* __restrict__ Out, int S, int N, int H, int GS) {
  __shared__ __align__(16) unsigned short Klds[2][32 * KPAD];   // [buf][key][h]
  __shared__ __align__(16) unsigned short Vtlds[2][64 * VPAD];  // [buf][h][key]
  __shared__ __align__(16) unsigned short Plds[8 * 16 * PPAD];  // per-wave P

  const int QB  = S / 128;
  const int bid = blockIdx.x;
  const int b   = bid / (N * QB);
  const int rem = bid % (N * QB);
  const int n   = rem / QB;
  const int q0  = (rem % QB) * 128;

  int st = 0, en = S;
#pragma unroll 1
  for (int s = 0; s < nseg; ++s) {
    int a = (int)modal_index[2 * s], e = (int)modal_index[2 * s + 1];
    if (q0 >= a && q0 < e) { st = a; en = e; }
  }

  const int tid  = threadIdx.x;
  const int wave = tid >> 5;
  const int lane = tid & 31;
  const int half = lane >> 4;
  const int l15  = lane & 15;
  const float scale = 0.125f;  // 1/sqrt(64)

  // Q fragments straight from global (A-matrix 16x32 bf16 layout).
  const int qrow = q0 + wave * 16 + l15;
  const float* qptr = Q + (((size_t)b * S + qrow) * N + n) * H;
  bf16x16 aq[2];
#pragma unroll
  for (int p2 = 0; p2 < 2; ++p2) {
#pragma unroll
    for (int g = 0; g < 2; ++g) {
      const float* p = qptr + half * 16 + p2 * 32 + g * 8;
      float4 f0 = *(const float4*)p;
      float4 f1 = *(const float4*)(p + 4);
      float t8[8] = {f0.x, f0.y, f0.z, f0.w, f1.x, f1.y, f1.z, f1.w};
#pragma unroll
      for (int i = 0; i < 8; ++i) aq[p2][g * 8 + i] = (__bf16)t8[i];
    }
  }

  bf16x16 ones;
#pragma unroll
  for (int i = 0; i < 16; ++i)
    ones[i] = __builtin_bit_cast(__bf16, (unsigned short)0x3F80);  // 1.0bf

  float m[8], l[8];
  v8f O[4];
#pragma unroll
  for (int r = 0; r < 8; ++r) { m[r] = -1e30f; l[r] = 0.0f; }
#pragma unroll
  for (int t = 0; t < 4; ++t)
#pragma unroll
    for (int r = 0; r < 8; ++r) O[t][r] = 0.0f;

  const int gblocks = (gflag[0] != 0) ? (GS / 32) : 0;
  const int lblocks = (en - st) / 32;
  const int nblocks = gblocks + lblocks;

  const size_t bn = (size_t)b * N + n;
  unsigned short* Pw = Plds + wave * 16 * PPAD;

  // Per-thread staging coordinates (constant over the loop).
  const int sk = tid >> 3, kg = tid & 7;  // K: 32 keys x 8 groups of 16 B
  const int vh = tid >> 2, vg = tid & 3;  // Vt: 64 h-rows x 4 groups of 16 B

  // Issue the two async copies for key-block `blk` into LDS buffer `buf`.
  auto issue_tile = [&](int buf, int blk) {
    const bool isg_ = blk < gblocks;
    const int k0_ = isg_ ? blk * 32 : st + (blk - gblocks) * 32;
    const unsigned short* kt_ = isg_ ? GKp + bn * GS * H + (size_t)k0_ * H
                                     : Kp + bn * S * H + (size_t)k0_ * H;
    const unsigned short* vt_ = isg_ ? GVp + bn * H * GS + k0_
                                     : Vp + bn * H * S + k0_;
    const int vs_ = isg_ ? GS : S;
    async_cp16(lds_off32(&Klds[buf][sk * KPAD + kg * 8]), kt_ + sk * H + kg * 8);
    async_cp16(lds_off32(&Vtlds[buf][vh * VPAD + vg * 8]), vt_ + (size_t)vh * vs_ + vg * 8);
  };

  if constexpr (PRE) {
    if (nblocks > 0) issue_tile(0, 0);  // prologue prefetch
  }

#pragma unroll 1
  for (int it = 0; it < nblocks; ++it) {
    const bool isg = it < gblocks;
    const int key0 = isg ? it * 32 : st + (it - gblocks) * 32;
    const float* mrow = isg ? gmask + (size_t)b * GS : lmask + (size_t)b * S;
    const int cur = it & 1;

    __syncthreads();  // all waves done reading the buffer we are about to refill
    if constexpr (PRE) {
      if (it + 1 < nblocks) {
        issue_tile(cur ^ 1, it + 1);  // prefetch next tile while computing this one
        async_wait2();                // my copies for `cur` (issued last iter) landed
      } else {
        async_wait0();
      }
    } else {
      const size_t ks = isg ? (size_t)H : (size_t)N * H;
      const float* kb = isg ? GK + (bn * GS + key0) * H
                            : K + (((size_t)b * S + key0) * N + n) * H;
      const float* vb = isg ? GV + (bn * GS + key0) * H
                            : V + (((size_t)b * S + key0) * N + n) * H;
      const float* ksrc = kb + sk * ks + kg * 8;
      const float* vsrc = vb + sk * ks + kg * 8;
      float4 a0 = *(const float4*)ksrc, a1 = *(const float4*)(ksrc + 4);
      float4 c0 = *(const float4*)vsrc, c1 = *(const float4*)(vsrc + 4);
      float ka[8] = {a0.x, a0.y, a0.z, a0.w, a1.x, a1.y, a1.z, a1.w};
      float va[8] = {c0.x, c0.y, c0.z, c0.w, c1.x, c1.y, c1.z, c1.w};
      *(u32x4*)&Klds[0][sk * KPAD + kg * 8] = pk8(ka);
#pragma unroll
      for (int i = 0; i < 8; ++i) Vtlds[0][(kg * 8 + i) * VPAD + sk] = bfbits(va[i]);
    }
    __syncthreads();  // everyone's staging for `cur` visible

    const unsigned short* Kc = PRE ? Klds[cur] : Klds[0];
    const unsigned short* Vc = PRE ? Vtlds[cur] : Vtlds[0];

    // ---- Scores: 4 WMMAs.
    v8f s0, s1;
#pragma unroll
    for (int r = 0; r < 8; ++r) { s0[r] = 0.0f; s1[r] = 0.0f; }
#pragma unroll
    for (int p2 = 0; p2 < 2; ++p2) {
      bf16x16 bk0 = frag2(&Kc[l15 * KPAD + p2 * 32 + half * 16], 0, 8);
      bf16x16 bk1 = frag2(&Kc[(16 + l15) * KPAD + p2 * 32 + half * 16], 0, 8);
      s0 = wmma_bf16(aq[p2], bk0, s0);
      s1 = wmma_bf16(aq[p2], bk1, s1);
    }

    const float mk0 = mrow[key0 + l15];
    const float mk1 = mrow[key0 + 16 + l15];

    // ---- Online softmax: max via permlane butterflies; sum via ones-WMMA.
    float alpha[8];
#pragma unroll
    for (int r = 0; r < 8; ++r) {
      float x0 = s0[r] * scale + mk0;
      float x1 = s1[r] * scale + mk1;
      float bm = rowmax16(fmaxf(x0, x1));
      float mn = fmaxf(m[r], bm);
      alpha[r] = __expf(m[r] - mn);
      m[r] = mn;
      float p0 = __expf(x0 - mn);
      float p1 = __expf(x1 - mn);
      const int row = r + half * 8;
      Pw[row * PPAD + l15]      = bfbits(p0);
      Pw[row * PPAD + 16 + l15] = bfbits(p1);
#pragma unroll
      for (int t = 0; t < 4; ++t) O[t][r] *= alpha[r];
    }

    // A-frag of P (D-layout -> A-layout via LDS round trip, same wave).
    bf16x16 ap = frag2(&Pw[l15 * PPAD + half * 16], 0, 8);

    // Row sums: P(16x32) x ones(32x16) -> every lane's element r = rowsum(r).
    v8f ls;
#pragma unroll
    for (int r = 0; r < 8; ++r) ls[r] = 0.0f;
    ls = wmma_bf16(ap, ones, ls);
#pragma unroll
    for (int r = 0; r < 8; ++r) l[r] = l[r] * alpha[r] + ls[r];

    // ---- PV: 4 WMMAs.
#pragma unroll
    for (int t = 0; t < 4; ++t) {
      bf16x16 bv = frag2(&Vc[(t * 16 + l15) * VPAD + half * 16], 0, 8);
      O[t] = wmma_bf16(ap, bv, O[t]);
    }
  }

  // ---- Epilogue.
#pragma unroll
  for (int t = 0; t < 4; ++t) {
#pragma unroll
    for (int r = 0; r < 8; ++r) {
      const int row = q0 + wave * 16 + r + half * 8;
      const int h = t * 16 + l15;
      Out[(((size_t)b * S + row) * N + n) * H + h] = O[t][r] / l[r];
    }
  }
}

extern "C" void kernel_launch(void* const* d_in, const int* in_sizes, int n_in,
                              void* d_out, int out_size, void* d_ws, size_t ws_size,
                              hipStream_t stream) {
  (void)n_in; (void)out_size;
  const long long* modal = (const long long*)d_in[0];
  const float* Q     = (const float*)d_in[1];
  const float* K     = (const float*)d_in[2];
  const float* V     = (const float*)d_in[3];
  const float* lmask = (const float*)d_in[4];
  const int*   gflag = (const int*)d_in[5];
  const float* GK    = (const float*)d_in[6];
  const float* GV    = (const float*)d_in[7];
  const float* gmask = (const float*)d_in[8];
  float* Out = (float*)d_out;

  const int B = 2, S = 4096, N = 16, H = 64, GS = 128;
  const int nseg = in_sizes[0] / 2;
  const int blocks = B * N * (S / 128);

  const size_t eK = (size_t)B * N * S * H;   // elements per converted tensor
  const size_t eG = (size_t)B * N * GS * H;
  const size_t need = (2 * eK + 2 * eG) * sizeof(unsigned short);

  if (ws_size >= need && d_ws) {
    unsigned short* Kp  = (unsigned short*)d_ws;
    unsigned short* Vp  = Kp + eK;
    unsigned short* GKp = Vp + eK;
    unsigned short* GVp = GKp + eG;
    // K (B,S,N,H) -> (B,N,S,H) bf16 ; GK already (B,N,GS,H)-ordered
    cvt_rows_kernel<<<B * N * (S / 32), 256, 0, stream>>>(
        K, Kp, S, (size_t)S * N * H, (size_t)H, (size_t)N * H, N);
    cvt_rows_kernel<<<B * N * (GS / 32), 256, 0, stream>>>(
        GK, GKp, GS, (size_t)N * GS * H, (size_t)GS * H, (size_t)H, N);
    // V (B,S,N,H) -> (B,N,H,S) bf16 ; GV (B,N,GS,H) -> (B,N,H,GS)
    tr64_kernel<<<B * N * (S / 64), 256, 0, stream>>>(
        V, Vp, S, (size_t)S * N * H, (size_t)H, (size_t)N * H, N);
    tr64_kernel<<<B * N * (GS / 64), 256, 0, stream>>>(
        GV, GVp, GS, (size_t)N * GS * H, (size_t)GS * H, (size_t)H, N);
    mmattn_fa_kernel<true><<<blocks, 256, 0, stream>>>(
        modal, nseg, Q, K, V, lmask, gflag, GK, GV, gmask,
        Kp, Vp, GKp, GVp, Out, S, N, H, GS);
  } else {
    mmattn_fa_kernel<false><<<blocks, 256, 0, stream>>>(
        modal, nseg, Q, K, V, lmask, gflag, GK, GV, gmask,
        nullptr, nullptr, nullptr, nullptr, Out, S, N, H, GS);
  }
}
